// MinGRU_53936199303793
// MI455X (gfx1250) — matrix operand order repaired and checked by reading, not compiled, not run on hardware
//
#include <hip/hip_runtime.h>
#include <hip/hip_bf16.h>

typedef __attribute__((ext_vector_type(16))) __bf16 v16bf;
typedef __attribute__((ext_vector_type(8)))  __bf16 v8bf;
typedef __attribute__((ext_vector_type(8)))  float  v8f;
typedef __attribute__((ext_vector_type(4)))  float  f32x4;

#define B_   256
#define T_   512
#define F_   256
#define H_   512
#define C_   10
#define KC   (H_ + F_)          /* 768 combined K dim: [h | xt] */
#define BT   16                 /* batch rows per workgroup (one WMMA M tile) */
#define KT   (KC / 32)          /* 24 K-tiles of 32 */
#define KTH  (H_ / 32)          /* 16 K-tiles covering the h part */
#define NTL  (H_ / 16)          /* 32 N-tiles of 16 */
#define WAVES   16
#define THREADS (WAVES * 32)

#define SA_STRIDE  (KC + 8)     /* 776 halves, 16B-aligned rows */
#define SRH_STRIDE (H_ + 8)     /* 520 halves */
#define SH_STRIDE  (H_ + 4)     /* 516 floats */

#define GATE_ELEMS (KT * NTL * 32 * 16)   /* 393216 bf16 per gate */

// ---------------------------------------------------------------------------
// Kernel 1: convert fp32 weights -> bf16, pre-swizzled into per-lane WMMA
// B-fragment order: dst[(((g*KT+kt)*NTL+nt)*32+lane)*16 + i]
//   = W_g[kt*32 + (lane>=16?8:0) + (i&7) + (i>=8?16:0)][nt*16 + (lane&15)]
// so each lane's B operand is 16 contiguous halves (32 bytes).
// ---------------------------------------------------------------------------
__global__ void gru_swizzle_weights(const float* __restrict__ Wz,
                                    const float* __restrict__ Wr,
                                    const float* __restrict__ Wh,
                                    __hip_bfloat16* __restrict__ dst) {
    int tid = blockIdx.x * blockDim.x + threadIdx.x;
    if (tid >= 3 * GATE_ELEMS) return;
    int g  = tid / GATE_ELEMS;
    int r0 = tid - g * GATE_ELEMS;
    int kt = r0 >> 14;             /* / (NTL*32*16 = 16384) */
    int r1 = r0 & 16383;
    int nt = r1 >> 9;              /* / 512 */
    int r2 = r1 & 511;
    int lane = r2 >> 4;
    int i    = r2 & 15;
    int k = kt * 32 + ((lane >> 4) << 3) + (i & 7) + ((i >> 3) << 4);
    int n = nt * 16 + (lane & 15);
    const float* W = (g == 0) ? Wz : (g == 1) ? Wr : Wh;
    dst[tid] = __float2bfloat16(W[k * H_ + n]);
}

// ---------------------------------------------------------------------------
// A-fragment load (16-bit A 16x32 layout): lane group picks K 0-7/8-15 plus
// K 16-23/24-31; p points at sBuf[row m][ktbase + (lane>>4)*8].
// ---------------------------------------------------------------------------
__device__ __forceinline__ v16bf load_a_frag(const __hip_bfloat16* p) {
    union { v16bf v; v8bf h[2]; } u;
    u.h[0] = *(const v8bf*)(const void*)(p);
    u.h[1] = *(const v8bf*)(const void*)(p + 16);
    return u.v;
}

__device__ __forceinline__ v16bf load_b_frag(const __hip_bfloat16* p) {
    return *(const v16bf*)(const void*)(p);
}

__device__ __forceinline__ float sigmoid_f(float x) {
    return 1.0f / (1.0f + __expf(-x));
}

// ---------------------------------------------------------------------------
// Kernel 2: persistent GRU. One workgroup owns 16 batch rows for all T steps.
// 16 waves; wave w computes output N-tiles {2w, 2w+1} for every gate.
// x_{t+1} is register-double-buffered (nontemporal loads issued at the start
// of step t, consumed at the top of step t+1) so HBM latency never sits on
// the serial inter-step critical path.
// ---------------------------------------------------------------------------
__global__ void __launch_bounds__(THREADS, 1)
gru_persistent(const float* __restrict__ x,
               const float* __restrict__ Wz_b,
               const float* __restrict__ Wr_b,
               const float* __restrict__ Wh_b,
               const float* __restrict__ fc_w,
               const float* __restrict__ fc_b,
               const __hip_bfloat16* __restrict__ Wswz,
               float* __restrict__ out) {
    __shared__ __hip_bfloat16 sA [BT][SA_STRIDE];   /* [h(bf16) | xt(bf16)] */
    __shared__ __hip_bfloat16 sRH[BT][SRH_STRIDE];  /* r * h (bf16)         */
    __shared__ float          sH [BT][SH_STRIDE];   /* h master (fp32)      */

    const int lane = threadIdx.x & 31;
    const int wid  = threadIdx.x >> 5;
    const int nt0  = wid * 2;
    const int nt1  = nt0 + 1;
    const int b0   = blockIdx.x * BT;

    /* C/D tile coords for this lane (8 rows per lane) */
    const int cm = (lane >> 4) << 3;     /* 0 or 8 */
    const int cn = lane & 15;
    /* A-fragment coords */
    const int am  = lane & 15;
    const int akb = (lane >> 4) << 3;    /* 0 or 8 */

    const int col0 = nt0 * 16 + cn;
    const int col1 = nt1 * 16 + cn;

    /* per-column biases (registers) */
    const float bzc0 = Wz_b[col0], bzc1 = Wz_b[col1];
    const float brc0 = Wr_b[col0], brc1 = Wr_b[col1];
    const float bhc0 = Wh_b[col0], bhc1 = Wh_b[col1];

    const __hip_bfloat16* Wz_s = Wswz + (size_t)lane * 16;
    const __hip_bfloat16* Wr_s = Wz_s + (size_t)GATE_ELEMS;
    const __hip_bfloat16* Wh_s = Wz_s + (size_t)2 * GATE_ELEMS;

    /* x prefetch coordinates: this thread owns two float4 chunks per step */
    const int v0   = threadIdx.x;            /* chunk ids in [0, 1024) */
    const int v1   = threadIdx.x + THREADS;
    const int row0 = v0 >> 6, f40 = v0 & 63; /* 64 float4 per row */
    const int row1 = v1 >> 6, f41 = v1 & 63;
    const float* xrow0 = x + (size_t)(b0 + row0) * T_ * F_ + f40 * 4;
    const float* xrow1 = x + (size_t)(b0 + row1) * T_ * F_ + f41 * 4;

    /* h = 0 */
    for (int v = threadIdx.x; v < BT * H_; v += THREADS) {
        int row = v >> 9, c = v & (H_ - 1);
        sH[row][c] = 0.0f;
        sA[row][c] = __float2bfloat16(0.0f);
    }

    /* prefetch x_0 (nontemporal: keep streamed x out of the weight-hot L2) */
    f32x4 pf0 = __builtin_nontemporal_load((const f32x4*)(const void*)(xrow0));
    f32x4 pf1 = __builtin_nontemporal_load((const f32x4*)(const void*)(xrow1));

    for (int t = 0; t < T_; ++t) {
        /* ---- commit prefetched x_t into sA[:, H_..H_+F_) as packed bf16 */
        {
            union { __hip_bfloat16 h[4]; uint2 u; } p;
            p.h[0] = __float2bfloat16(pf0[0]); p.h[1] = __float2bfloat16(pf0[1]);
            p.h[2] = __float2bfloat16(pf0[2]); p.h[3] = __float2bfloat16(pf0[3]);
            *(uint2*)(void*)&sA[row0][H_ + f40 * 4] = p.u;
            p.h[0] = __float2bfloat16(pf1[0]); p.h[1] = __float2bfloat16(pf1[1]);
            p.h[2] = __float2bfloat16(pf1[2]); p.h[3] = __float2bfloat16(pf1[3]);
            *(uint2*)(void*)&sA[row1][H_ + f41 * 4] = p.u;
        }
        __syncthreads();

        /* ---- issue x_{t+1} loads now; they retire under the GEMMs ---- */
        if (t + 1 < T_) {
            pf0 = __builtin_nontemporal_load(
                      (const f32x4*)(const void*)(xrow0 + (size_t)(t + 1) * F_));
            pf1 = __builtin_nontemporal_load(
                      (const f32x4*)(const void*)(xrow1 + (size_t)(t + 1) * F_));
        }

        /* ---- z and r GEMMs: [16,768] x [768, 2x16] each ---- */
        v8f az0 = {}, az1 = {}, ar0 = {}, ar1 = {};
#pragma unroll 4
        for (int kt = 0; kt < KT; ++kt) {
            v16bf a = load_a_frag(&sA[am][kt * 32 + akb]);
            size_t base = (size_t)kt * NTL * 512;
            v16bf bz0 = load_b_frag(Wz_s + base + (size_t)nt0 * 512);
            v16bf bz1 = load_b_frag(Wz_s + base + (size_t)nt1 * 512);
            v16bf br0 = load_b_frag(Wr_s + base + (size_t)nt0 * 512);
            v16bf br1 = load_b_frag(Wr_s + base + (size_t)nt1 * 512);
            az0 = __builtin_amdgcn_wmma_f32_16x16x32_bf16(false, a, false, bz0,
                                                          (short)0, az0, false, false);
            az1 = __builtin_amdgcn_wmma_f32_16x16x32_bf16(false, a, false, bz1,
                                                          (short)0, az1, false, false);
            ar0 = __builtin_amdgcn_wmma_f32_16x16x32_bf16(false, a, false, br0,
                                                          (short)0, ar0, false, false);
            ar1 = __builtin_amdgcn_wmma_f32_16x16x32_bf16(false, a, false, br1,
                                                          (short)0, ar1, false, false);
        }

        /* z kept in registers (this wave owns the same columns for h~);
           r*h written to LDS (needed by all waves along K). */
        float zv0[8], zv1[8];
#pragma unroll
        for (int j = 0; j < 8; ++j) {
            int m = cm + j;
            float z0 = sigmoid_f(az0[j] + bzc0);
            float z1 = sigmoid_f(az1[j] + bzc1);
            zv0[j] = z0; zv1[j] = z1;
            float r0 = sigmoid_f(ar0[j] + brc0);
            float r1 = sigmoid_f(ar1[j] + brc1);
            sRH[m][col0] = __float2bfloat16(r0 * sH[m][col0]);
            sRH[m][col1] = __float2bfloat16(r1 * sH[m][col1]);
        }
        __syncthreads();

        /* ---- h~ GEMM: A = [r*h | xt] ---- */
        v8f ah0 = {}, ah1 = {};
#pragma unroll 4
        for (int kt = 0; kt < KT; ++kt) {
            const __hip_bfloat16* ap = (kt < KTH)
                ? &sRH[am][kt * 32 + akb]
                : &sA [am][kt * 32 + akb];   /* kt*32 >= 512 lands in xt region */
            v16bf a = load_a_frag(ap);
            size_t base = (size_t)kt * NTL * 512;
            v16bf bh0 = load_b_frag(Wh_s + base + (size_t)nt0 * 512);
            v16bf bh1 = load_b_frag(Wh_s + base + (size_t)nt1 * 512);
            ah0 = __builtin_amdgcn_wmma_f32_16x16x32_bf16(false, a, false, bh0,
                                                          (short)0, ah0, false, false);
            ah1 = __builtin_amdgcn_wmma_f32_16x16x32_bf16(false, a, false, bh1,
                                                          (short)0, ah1, false, false);
        }

        /* ---- h_new = (1-z)*h + z*tanh(.) ; update fp32 master + bf16 copy */
#pragma unroll
        for (int j = 0; j < 8; ++j) {
            int m = cm + j;
            {
                float ht = tanhf(ah0[j] + bhc0);
                float h  = sH[m][col0];
                float hn = (1.0f - zv0[j]) * h + zv0[j] * ht;
                sH[m][col0] = hn;
                sA[m][col0] = __float2bfloat16(hn);
            }
            {
                float ht = tanhf(ah1[j] + bhc1);
                float h  = sH[m][col1];
                float hn = (1.0f - zv1[j]) * h + zv1[j] * ht;
                sH[m][col1] = hn;
                sA[m][col1] = __float2bfloat16(hn);
            }
        }
        __syncthreads();
    }

    /* ---- final FC: [16,512] x [512,10] + bias, tiny, once ---- */
    if (threadIdx.x < BT * C_) {
        int row = threadIdx.x / C_;
        int c   = threadIdx.x % C_;
        float acc = fc_b[c];
        for (int j = 0; j < H_; ++j)
            acc += sH[row][j] * fc_w[j * C_ + c];
        out[(size_t)(b0 + row) * C_ + c] = acc;
    }
}

extern "C" void kernel_launch(void* const* d_in, const int* in_sizes, int n_in,
                              void* d_out, int out_size, void* d_ws, size_t ws_size,
                              hipStream_t stream) {
    const float* x    = (const float*)d_in[0];
    const float* Wz_w = (const float*)d_in[1];
    const float* Wz_b = (const float*)d_in[2];
    const float* Wr_w = (const float*)d_in[3];
    const float* Wr_b = (const float*)d_in[4];
    const float* Wh_w = (const float*)d_in[5];
    const float* Wh_b = (const float*)d_in[6];
    const float* fc_w = (const float*)d_in[7];
    const float* fc_b = (const float*)d_in[8];
    __hip_bfloat16* Wswz = (__hip_bfloat16*)d_ws;   /* 3 * 768KB bf16 */

    const int total = 3 * GATE_ELEMS;
    gru_swizzle_weights<<<(total + 255) / 256, 256, 0, stream>>>(Wz_w, Wr_w, Wh_w, Wswz);
    gru_persistent<<<B_ / BT, THREADS, 0, stream>>>(x, Wz_b, Wr_b, Wh_b,
                                                    fc_w, fc_b, Wswz, (float*)d_out);
}
